// TransformerBlock_42417097015909
// MI455X (gfx1250) — compile-verified, matrix operand order checked by reading, not run on hardware
//
#include <hip/hip_runtime.h>
#include <math.h>

// ---------------------------------------------------------------------------
// CDNA5 (gfx1250) TransformerBlock: RMSNorm + relative attention + FFN.
// All dense contractions go through v_wmma_f32_16x16x32_f16 (wave32 WMMA).
// Tile staging uses global_load_async_to_lds_b128 (double-buffered pipeline).
// ---------------------------------------------------------------------------

typedef __attribute__((ext_vector_type(16))) _Float16 v16h;
typedef __attribute__((ext_vector_type(8)))  float    v8f;
typedef __attribute__((ext_vector_type(4)))  unsigned int u32x4;
typedef __attribute__((ext_vector_type(4)))  int          v4i;

union Frag16 { v16h v; u32x4 u[2]; };
union H8 { u32x4 u; _Float16 h[8]; };

#define WMMA_F16(ACC, AF, BF)                                                  \
  (ACC) = __builtin_amdgcn_wmma_f32_16x16x32_f16(false, (AF), false, (BF),     \
                                                 (short)0, (ACC), false, false)

#if defined(__has_builtin)
#if __has_builtin(__builtin_amdgcn_global_load_async_to_lds_b128) &&           \
    __has_builtin(__builtin_amdgcn_s_wait_asynccnt)
#define ATHENA_ASYNC_LDS 1
#endif
#endif

#ifdef ATHENA_ASYNC_LDS
#define ASYNC_CP16(dstLds, srcGlb)                                             \
  __builtin_amdgcn_global_load_async_to_lds_b128(                              \
      (__attribute__((address_space(1))) v4i*)(srcGlb),                        \
      (__attribute__((address_space(3))) v4i*)(dstLds), 0, 0)
#define ASYNC_WAIT() __builtin_amdgcn_s_wait_asynccnt(0)
#else
#define ASYNC_CP16(dstLds, srcGlb) (*(u32x4*)(dstLds) = *(const u32x4*)(srcGlb))
#define ASYNC_WAIT() ((void)0)
#endif

__device__ inline v8f v8f_zero() {
  v8f z = {0.f, 0.f, 0.f, 0.f, 0.f, 0.f, 0.f, 0.f};
  return z;
}

// ---------------------------------------------------------------------------
// f16 WMMA GEMM:  C[M,N] = A[M,K] * W[N,K]^T (+bias)(+gelu)(+resid)
// 256 threads (8 waves), 128x128 block tile, BK=32, double-buffered async
// LDS staging. Waves: 4(M) x 2(N); each wave owns 32x64 = 2x4 WMMA tiles.
// out_sbd: remap output rows token-major -> [S=1024, B=4, D] layout.
// ---------------------------------------------------------------------------
__global__ __launch_bounds__(256) void gemm_f16_wmma(
    const _Float16* __restrict__ A, const _Float16* __restrict__ W,
    const float* __restrict__ bias, const float* __restrict__ resid,
    float* __restrict__ Cf, _Float16* __restrict__ Ch,
    int M, int N, int K, int act, int out_sbd) {
  __shared__ __align__(16) _Float16 As[2][128][40];
  __shared__ __align__(16) _Float16 Bs[2][128][40];
  const int tid  = threadIdx.x;
  const int wave = tid >> 5, lane = tid & 31, lr = lane & 15;
  const int kb   = (lane < 16) ? 0 : 8;   // K sub-offset per A/B layout
  const int rofs = (lane < 16) ? 0 : 8;   // C/D row offset
  const int m0 = blockIdx.y * 128, n0 = blockIdx.x * 128;
  const int wm = (wave >> 1) * 32, wn = (wave & 1) * 64;

  // Straight-line staging: each thread owns 2 A-chunks + 2 B-chunks of 16B.
  const int ldr = tid >> 2;             // 0..63
  const int ldc = (tid & 3) << 3;       // 0,8,16,24
  const _Float16* gA0 = A + (size_t)(m0 + ldr) * K + ldc;
  const _Float16* gA1 = A + (size_t)(m0 + 64 + ldr) * K + ldc;
  const _Float16* gB0 = W + (size_t)(n0 + ldr) * K + ldc;
  const _Float16* gB1 = W + (size_t)(n0 + 64 + ldr) * K + ldc;
  _Float16* sA0[2] = {&As[0][ldr][ldc], &As[1][ldr][ldc]};
  _Float16* sA1[2] = {&As[0][64 + ldr][ldc], &As[1][64 + ldr][ldc]};
  _Float16* sB0[2] = {&Bs[0][ldr][ldc], &Bs[1][ldr][ldc]};
  _Float16* sB1[2] = {&Bs[0][64 + ldr][ldc], &Bs[1][64 + ldr][ldc]};

  v8f acc[2][4];
  for (int i = 0; i < 2; ++i)
    for (int j = 0; j < 4; ++j) acc[i][j] = v8f_zero();

  const int nk = K >> 5;
  // Prologue: stage K-step 0 into buffer 0.
  ASYNC_CP16(sA0[0], gA0); ASYNC_CP16(sA1[0], gA1);
  ASYNC_CP16(sB0[0], gB0); ASYNC_CP16(sB1[0], gB1);
  gA0 += 32; gA1 += 32; gB0 += 32; gB1 += 32;
  ASYNC_WAIT();
  __syncthreads();

  for (int kt = 0; kt < nk; ++kt) {
    const int cur = kt & 1, nxt = cur ^ 1;
    // Prefetch next K-step into the other buffer; overlaps the WMMAs below.
    if (kt + 1 < nk) {
      ASYNC_CP16(sA0[nxt], gA0); ASYNC_CP16(sA1[nxt], gA1);
      ASYNC_CP16(sB0[nxt], gB0); ASYNC_CP16(sB1[nxt], gB1);
      gA0 += 32; gA1 += 32; gB0 += 32; gB1 += 32;
    }

    Frag16 af[2], bf[4];
#pragma unroll
    for (int i = 0; i < 2; ++i) {
      af[i].u[0] = *(const u32x4*)&As[cur][wm + i * 16 + lr][kb];
      af[i].u[1] = *(const u32x4*)&As[cur][wm + i * 16 + lr][kb + 16];
    }
#pragma unroll
    for (int j = 0; j < 4; ++j) {
      bf[j].u[0] = *(const u32x4*)&Bs[cur][wn + j * 16 + lr][kb];
      bf[j].u[1] = *(const u32x4*)&Bs[cur][wn + j * 16 + lr][kb + 16];
    }
#pragma unroll
    for (int i = 0; i < 2; ++i)
#pragma unroll
      for (int j = 0; j < 4; ++j) WMMA_F16(acc[i][j], af[i].v, bf[j].v);

    ASYNC_WAIT();
    __syncthreads();
  }

#pragma unroll
  for (int i = 0; i < 2; ++i)
#pragma unroll
    for (int j = 0; j < 4; ++j)
#pragma unroll
      for (int r = 0; r < 8; ++r) {
        const int row = m0 + wm + i * 16 + rofs + r;
        const int col = n0 + wn + j * 16 + lr;
        float v = acc[i][j][r];
        if (bias) v += bias[col];
        if (act == 1) v = 0.5f * v * (1.0f + erff(v * 0.70710678118654752f));
        if (resid) v += resid[(size_t)row * N + col];
        size_t orow = (size_t)row;
        if (out_sbd) orow = (size_t)(row & 1023) * 4 + (row >> 10);
        if (Cf) Cf[orow * N + col] = v;
        if (Ch) Ch[orow * N + col] = (_Float16)v;
      }
}

// ---------------------------------------------------------------------------
// Relative attention, flash-style. Grid: (B*H, S/64). 4 waves x 16 q-rows.
//   scores = (Q.K^T + qrel[bin]) / 8 ; online softmax over 32-key tiles;
//   out = P.V (WMMA) + bins(P).rel_v (WMMA over 101->128 padded bins).
// ---------------------------------------------------------------------------
__global__ __launch_bounds__(128) void attn_rel_wmma(
    const _Float16* __restrict__ Q, const _Float16* __restrict__ Km,
    const _Float16* __restrict__ V, const _Float16* __restrict__ relk,
    const _Float16* __restrict__ relvT, _Float16* __restrict__ ctx) {
  const int S = 1024, D = 1024;
  const int bh = blockIdx.x, b = bh >> 4, h = bh & 15;
  const int q0 = blockIdx.y * 64;
  const int tid = threadIdx.x, wave = tid >> 5, lane = tid & 31, lr = lane & 15;
  const int kb = (lane < 16) ? 0 : 8;
  const int rofs = (lane < 16) ? 0 : 8;

  // Dynamic LDS carve (57,856 B total):
  extern __shared__ __align__(16) char smem_raw[];
  _Float16* KT = (_Float16*)smem_raw;   // [32][72]  keys x HD
  _Float16* VT = KT + 32 * 72;          // [64][40]  HD x keys (transposed V)
  _Float16* QR = VT + 64 * 40;          // [4][16][112]  q . rel_k table
  _Float16* PS = QR + 4 * 16 * 112;     // [4][16][40]   P staged as f16
  _Float16* RS = QR;                    // [4][16][128]  aliases QR+PS (dead)
  float* RB = (float*)(PS + 4 * 16 * 40);  // [4][16][112] prob bins (f32)

  const _Float16* kbase = Km + (size_t)(b * S) * D + h * 64;
  const _Float16* vbase = V + (size_t)(b * S) * D + h * 64;

  // --- Q fragments for this wave's 16 query rows (HD=64 -> 2 K-steps) ---
  const int qrow = q0 + wave * 16 + lr;
  Frag16 aq[2];
#pragma unroll
  for (int ks = 0; ks < 2; ++ks) {
    const _Float16* p = &Q[(size_t)(b * S + qrow) * D + h * 64 + ks * 32];
    aq[ks].u[0] = *(const u32x4*)(p + kb);
    aq[ks].u[1] = *(const u32x4*)(p + kb + 16);
  }

  // --- qrel = Q . rel_k^T  (rel-bin dim padded to 112) ---
#pragma unroll
  for (int rt = 0; rt < 7; ++rt) {
    v8f accq = v8f_zero();
#pragma unroll
    for (int ks = 0; ks < 2; ++ks) {
      Frag16 bfq;
      const _Float16* p = &relk[(size_t)(rt * 16 + lr) * 64 + ks * 32];
      bfq.u[0] = *(const u32x4*)(p + kb);
      bfq.u[1] = *(const u32x4*)(p + kb + 16);
      WMMA_F16(accq, aq[ks].v, bfq.v);
    }
#pragma unroll
    for (int r = 0; r < 8; ++r)
      QR[(wave * 16 + rofs + r) * 112 + rt * 16 + lr] = (_Float16)accq[r];
  }

  // --- zero the bin accumulators, init online-softmax state ---
#pragma unroll
  for (int r = 0; r < 8; ++r)
    for (int c = lr; c < 112; c += 16)
      RB[(wave * 16 + rofs + r) * 112 + c] = 0.f;

  float mrow[8], lrow[8];
#pragma unroll
  for (int r = 0; r < 8; ++r) { mrow[r] = -1e30f; lrow[r] = 0.f; }
  v8f acco[4];
#pragma unroll
  for (int j = 0; j < 4; ++j) acco[j] = v8f_zero();

  // Staging chunk assignments (straight-line, 128 threads):
  const int ck0 = tid, ck1 = tid + 128;         // 256 chunks of 8 halves
  const int key0 = ck0 >> 3, col0 = (ck0 & 7) << 3;
  const int key1 = ck1 >> 3, col1 = (ck1 & 7) << 3;

  // --- main loop over 32-key tiles ---
  for (int kt = 0; kt < 32; ++kt) {
    __syncthreads();
    // K tile [key][hd] via (async) b128 copies
    ASYNC_CP16(&KT[key0 * 72 + col0],
               kbase + (size_t)(kt * 32 + key0) * D + col0);
    ASYNC_CP16(&KT[key1 * 72 + col1],
               kbase + (size_t)(kt * 32 + key1) * D + col1);
    // V tile transposed [hd][key]: 2 vector loads + 16 scalar LDS stores
    {
      H8 rv0, rv1;
      rv0.u = *(const u32x4*)(vbase + (size_t)(kt * 32 + key0) * D + col0);
      rv1.u = *(const u32x4*)(vbase + (size_t)(kt * 32 + key1) * D + col1);
#pragma unroll
      for (int j = 0; j < 8; ++j) {
        VT[(col0 + j) * 40 + key0] = rv0.h[j];
        VT[(col1 + j) * 40 + key1] = rv1.h[j];
      }
    }
    ASYNC_WAIT();
    __syncthreads();

    // scores: Q[16x64] . K^T -> two 16x16 tiles
    v8f sc[2];
#pragma unroll
    for (int sub = 0; sub < 2; ++sub) {
      sc[sub] = v8f_zero();
#pragma unroll
      for (int ks = 0; ks < 2; ++ks) {
        Frag16 bfk;
        const _Float16* p = &KT[(sub * 16 + lr) * 72 + ks * 32];
        bfk.u[0] = *(const u32x4*)(p + kb);
        bfk.u[1] = *(const u32x4*)(p + kb + 16);
        WMMA_F16(sc[sub], aq[ks].v, bfk.v);
      }
    }

    // relative term + scale + online softmax (per-row over half-wave groups)
    float alpha[8];
#pragma unroll
    for (int r = 0; r < 8; ++r) {
      const int row = rofs + r;
      const int qq = q0 + wave * 16 + row;
      float vmax = -1e30f;
#pragma unroll
      for (int sub = 0; sub < 2; ++sub) {
        const int kk = kt * 32 + sub * 16 + lr;
        int dlt = kk - qq;
        dlt = dlt < -50 ? -50 : (dlt > 50 ? 50 : dlt);
        const float vsc =
            (sc[sub][r] + (float)QR[(wave * 16 + row) * 112 + dlt + 50]) * 0.125f;
        sc[sub][r] = vsc;
        vmax = fmaxf(vmax, vsc);
      }
#pragma unroll
      for (int o = 1; o < 16; o <<= 1) vmax = fmaxf(vmax, __shfl_xor(vmax, o, 32));
      const float mnew = fmaxf(mrow[r], vmax);
      alpha[r] = __expf(mrow[r] - mnew);
      mrow[r] = mnew;
      float rsum = 0.f;
#pragma unroll
      for (int sub = 0; sub < 2; ++sub) {
        const float p = __expf(sc[sub][r] - mnew);
        sc[sub][r] = p;
        rsum += p;
      }
#pragma unroll
      for (int o = 1; o < 16; o <<= 1) rsum += __shfl_xor(rsum, o, 32);
      lrow[r] = lrow[r] * alpha[r] + rsum;
    }

    // rescale running accumulators only when the max actually moved
    float amin = 1.f;
#pragma unroll
    for (int r = 0; r < 8; ++r) amin = fminf(amin, alpha[r]);
    if (amin < 1.f) {
#pragma unroll
      for (int j = 0; j < 4; ++j)
#pragma unroll
        for (int r = 0; r < 8; ++r) acco[j][r] *= alpha[r];
#pragma unroll
      for (int r = 0; r < 8; ++r) {
        const int row = rofs + r;
        for (int c = lr; c < 112; c += 16)
          RB[(wave * 16 + row) * 112 + c] *= alpha[r];
      }
    }

    // bin the probabilities (for attn.rel_v) and stage P as f16
#pragma unroll
    for (int r = 0; r < 8; ++r) {
      const int row = rofs + r;
      const int qq = q0 + wave * 16 + row;
#pragma unroll
      for (int sub = 0; sub < 2; ++sub) {
        const int kk = kt * 32 + sub * 16 + lr;
        int dlt = kk - qq;
        dlt = dlt < -50 ? -50 : (dlt > 50 ? 50 : dlt);
        atomicAdd(&RB[(wave * 16 + row) * 112 + dlt + 50], sc[sub][r]);
        PS[(wave * 16 + row) * 40 + sub * 16 + lr] = (_Float16)sc[sub][r];
      }
    }
    __syncthreads();

    // out += P[16x32] . V[32x64]
    Frag16 pa;
    pa.u[0] = *(const u32x4*)&PS[(wave * 16 + lr) * 40 + kb];
    pa.u[1] = *(const u32x4*)&PS[(wave * 16 + lr) * 40 + kb + 16];
#pragma unroll
    for (int j = 0; j < 4; ++j) {
      Frag16 bfv;
      const _Float16* p = &VT[(j * 16 + lr) * 40];
      bfv.u[0] = *(const u32x4*)(p + kb);
      bfv.u[1] = *(const u32x4*)(p + kb + 16);
      WMMA_F16(acco[j], pa.v, bfv.v);
    }
  }
  __syncthreads();

  // --- finalize: normalize, then out += bins . rel_v (K padded to 128) ---
  float invl[8];
#pragma unroll
  for (int r = 0; r < 8; ++r) invl[r] = 1.f / lrow[r];
#pragma unroll
  for (int j = 0; j < 4; ++j)
#pragma unroll
    for (int r = 0; r < 8; ++r) acco[j][r] *= invl[r];
#pragma unroll
  for (int r = 0; r < 8; ++r) {
    const int row = rofs + r;
    for (int c = lr; c < 112; c += 16)
      RS[(wave * 16 + row) * 128 + c] =
          (_Float16)(RB[(wave * 16 + row) * 112 + c] * invl[r]);
    for (int c = 112 + lr; c < 128; c += 16)
      RS[(wave * 16 + row) * 128 + c] = (_Float16)0.f;
  }
  __syncthreads();

#pragma unroll
  for (int ks = 0; ks < 4; ++ks) {
    Frag16 ar;
    const _Float16* p = &RS[(wave * 16 + lr) * 128 + ks * 32];
    ar.u[0] = *(const u32x4*)(p + kb);
    ar.u[1] = *(const u32x4*)(p + kb + 16);
#pragma unroll
    for (int j = 0; j < 4; ++j) {
      Frag16 br;
      const _Float16* q = &relvT[(size_t)(j * 16 + lr) * 128 + ks * 32];
      br.u[0] = *(const u32x4*)(q + kb);
      br.u[1] = *(const u32x4*)(q + kb + 16);
      WMMA_F16(acco[j], ar.v, br.v);
    }
  }

#pragma unroll
  for (int j = 0; j < 4; ++j)
#pragma unroll
    for (int r = 0; r < 8; ++r) {
      const int row = q0 + wave * 16 + rofs + r;
      const int dcol = h * 64 + j * 16 + lr;
      ctx[(size_t)(b * S + row) * D + dcol] = (_Float16)acco[j][r];
    }
}

// ---------------------------------------------------------------------------
// RMSNorm (one block per token). Optional [S,B,D] -> token-major remap.
// ---------------------------------------------------------------------------
__global__ __launch_bounds__(256) void rmsnorm_cast(
    const float* __restrict__ src, const float* __restrict__ g,
    float* __restrict__ outf, _Float16* __restrict__ outh,
    int Dd, int S, int Bsz, int sbd) {
  const int t = blockIdx.x;
  const size_t base =
      sbd ? (size_t)((t % S) * Bsz + (t / S)) * Dd : (size_t)t * Dd;
  __shared__ float red[256];
  float ss = 0.f;
  for (int d = threadIdx.x; d < Dd; d += 256) {
    const float x = src[base + d];
    ss += x * x;
  }
  red[threadIdx.x] = ss;
  __syncthreads();
  for (int s = 128; s > 0; s >>= 1) {
    if (threadIdx.x < s) red[threadIdx.x] += red[threadIdx.x + s];
    __syncthreads();
  }
  const float rms = sqrtf(red[0]) * rsqrtf((float)Dd);
  const float inv = 1.f / (rms + 1e-8f);
  for (int d = threadIdx.x; d < Dd; d += 256) {
    const float y = g[d] * src[base + d] * inv;
    if (outf) outf[(size_t)t * Dd + d] = y;
    if (outh) outh[(size_t)t * Dd + d] = (_Float16)y;
  }
}

__global__ void cvt_f32_f16(const float* __restrict__ s,
                            _Float16* __restrict__ d, size_t n) {
  const size_t i = (size_t)blockIdx.x * blockDim.x + threadIdx.x;
  if (i < n) d[i] = (_Float16)s[i];
}

// rel_k -> f16 [128][64] (rows >=101 zero); rel_v -> f16 transposed [64][128].
__global__ void build_rel_tables(const float* __restrict__ rk,
                                 const float* __restrict__ rv,
                                 _Float16* __restrict__ rk16,
                                 _Float16* __restrict__ rvT) {
  const int i = blockIdx.x * blockDim.x + threadIdx.x;
  if (i < 128 * 64) {
    const int r = i >> 6, d = i & 63;
    const float kv = (r < 101) ? rk[r * 64 + d] : 0.f;
    const float vv = (r < 101) ? rv[r * 64 + d] : 0.f;
    rk16[r * 64 + d] = (_Float16)kv;
    rvT[d * 128 + r] = (_Float16)vv;
  }
}

// ---------------------------------------------------------------------------
extern "C" void kernel_launch(void* const* d_in, const int* in_sizes, int n_in,
                              void* d_out, int out_size, void* d_ws,
                              size_t ws_size, hipStream_t stream) {
  (void)in_sizes; (void)n_in; (void)out_size; (void)ws_size;
  const int S = 1024, B = 4, D = 1024, T = S * B, DF = 4 * D;

  const float* x      = (const float*)d_in[0];
  const float* Wq     = (const float*)d_in[1];
  const float* bq     = (const float*)d_in[2];
  const float* Wk     = (const float*)d_in[3];
  const float* bk     = (const float*)d_in[4];
  const float* Wv     = (const float*)d_in[5];
  const float* bv     = (const float*)d_in[6];
  const float* Wo     = (const float*)d_in[7];
  const float* bo     = (const float*)d_in[8];
  const float* rel_k  = (const float*)d_in[9];
  const float* rel_v  = (const float*)d_in[10];
  const float* g_attn = (const float*)d_in[11];
  const float* g_ff   = (const float*)d_in[12];
  const float* W_in   = (const float*)d_in[13];
  const float* b_in   = (const float*)d_in[14];
  const float* W_out  = (const float*)d_in[15];
  const float* b_out  = (const float*)d_in[16];

  char* ws = (char*)d_ws;
  size_t off = 0;
  auto alloc = [&](size_t bytes) {
    void* p = ws + off;
    off = (off + bytes + 255) & ~(size_t)255;
    return p;
  };

  float*    hnf    = (float*)   alloc((size_t)T * D * 4);
  _Float16* hnh    = (_Float16*)alloc((size_t)T * D * 2);
  _Float16* q16    = (_Float16*)alloc((size_t)T * D * 2);
  _Float16* k16    = (_Float16*)alloc((size_t)T * D * 2);
  _Float16* v16    = (_Float16*)alloc((size_t)T * D * 2);
  _Float16* ctx16  = (_Float16*)alloc((size_t)T * D * 2);
  float*    h2     = (float*)   alloc((size_t)T * D * 4);
  _Float16* ff16   = (_Float16*)alloc((size_t)T * D * 2);
  _Float16* ffa16  = (_Float16*)alloc((size_t)T * DF * 2);
  _Float16* wq16   = (_Float16*)alloc((size_t)D * D * 2);
  _Float16* wk16   = (_Float16*)alloc((size_t)D * D * 2);
  _Float16* wv16   = (_Float16*)alloc((size_t)D * D * 2);
  _Float16* wo16   = (_Float16*)alloc((size_t)D * D * 2);
  _Float16* win16  = (_Float16*)alloc((size_t)DF * D * 2);
  _Float16* wout16 = (_Float16*)alloc((size_t)D * DF * 2);
  _Float16* rk16   = (_Float16*)alloc((size_t)128 * 64 * 2);
  _Float16* rvT16  = (_Float16*)alloc((size_t)64 * 128 * 2);

  const size_t nDD = (size_t)D * D, nDF = (size_t)DF * D;
  cvt_f32_f16<<<(int)((nDD + 255) / 256), 256, 0, stream>>>(Wq, wq16, nDD);
  cvt_f32_f16<<<(int)((nDD + 255) / 256), 256, 0, stream>>>(Wk, wk16, nDD);
  cvt_f32_f16<<<(int)((nDD + 255) / 256), 256, 0, stream>>>(Wv, wv16, nDD);
  cvt_f32_f16<<<(int)((nDD + 255) / 256), 256, 0, stream>>>(Wo, wo16, nDD);
  cvt_f32_f16<<<(int)((nDF + 255) / 256), 256, 0, stream>>>(W_in, win16, nDF);
  cvt_f32_f16<<<(int)((nDF + 255) / 256), 256, 0, stream>>>(W_out, wout16, nDF);
  build_rel_tables<<<(128 * 64 + 255) / 256, 256, 0, stream>>>(rel_k, rel_v,
                                                               rk16, rvT16);

  // pre-norm (also remaps [S,B,D] -> token-major)
  rmsnorm_cast<<<T, 256, 0, stream>>>(x, g_attn, hnf, hnh, D, S, B, 1);

  const dim3 gblk(256);
  // Q/K/V projections (f16 out for attention)
  gemm_f16_wmma<<<dim3(D / 128, T / 128), gblk, 0, stream>>>(
      hnh, wq16, bq, nullptr, nullptr, q16, T, D, D, 0, 0);
  gemm_f16_wmma<<<dim3(D / 128, T / 128), gblk, 0, stream>>>(
      hnh, wk16, bk, nullptr, nullptr, k16, T, D, D, 0, 0);
  gemm_f16_wmma<<<dim3(D / 128, T / 128), gblk, 0, stream>>>(
      hnh, wv16, bv, nullptr, nullptr, v16, T, D, D, 0, 0);

  // relative attention
  const size_t attn_lds =
      (size_t)(32 * 72 + 64 * 40 + 4 * 16 * 112 + 4 * 16 * 40) * 2 +
      (size_t)4 * 16 * 112 * 4;  // 57,856 B
  attn_rel_wmma<<<dim3(B * 16, S / 64), dim3(128), attn_lds, stream>>>(
      q16, k16, v16, rk16, rvT16, ctx16);

  // output projection + residual onto pre-normed h
  gemm_f16_wmma<<<dim3(D / 128, T / 128), gblk, 0, stream>>>(
      ctx16, wo16, bo, hnf, h2, nullptr, T, D, D, 0, 0);

  // FFN
  rmsnorm_cast<<<T, 256, 0, stream>>>(h2, g_ff, nullptr, ff16, D, S, B, 0);
  gemm_f16_wmma<<<dim3(DF / 128, T / 128), gblk, 0, stream>>>(
      ff16, win16, b_in, nullptr, nullptr, ffa16, T, DF, D, 1, 0);
  // final projection + residual, written straight out in [S,B,D] order
  gemm_f16_wmma<<<dim3(D / 128, T / 128), gblk, 0, stream>>>(
      ffa16, wout16, b_out, h2, (float*)d_out, nullptr, T, D, DF, 0, 1);
}
